// RNNDecoder_78262894068501
// MI455X (gfx1250) — compile-verified, matrix operand order checked by reading, not run on hardware
//
#include <hip/hip_runtime.h>

typedef __bf16 bf16x16 __attribute__((ext_vector_type(16)));
typedef float v8f __attribute__((ext_vector_type(8)));
typedef unsigned v4u __attribute__((ext_vector_type(4)));
typedef int v8i __attribute__((ext_vector_type(8)));
typedef int v4i __attribute__((ext_vector_type(4)));

#if defined(__HIP_DEVICE_COMPILE__) && defined(__gfx1250__) && \
    __has_builtin(__builtin_amdgcn_tensor_load_to_lds) && \
    __has_builtin(__builtin_amdgcn_s_wait_tensorcnt)
#define USE_TDM 1
#else
#define USE_TDM 0
#endif

union FragBF { bf16x16 v; uint4 q[2]; };

__device__ inline v8f bfwmma(const FragBF& a, const FragBF& b, v8f c) {
    return __builtin_amdgcn_wmma_f32_16x16x32_bf16(false, a.v, false, b.v,
                                                   (short)0, c, false, false);
}

__device__ inline unsigned short f2bf(float f) {
    union { float f; unsigned u; } v; v.f = f;
    unsigned r = v.u + 0x7FFFu + ((v.u >> 16) & 1u);
    return (unsigned short)(r >> 16);
}

__device__ inline float sigmoidf(float x) { return 1.f / (1.f + __expf(-x)); }

// ---------------------------------------------------------------- utilities
__global__ void k_zero(uint4* p, long n) {
    long i = (long)blockIdx.x * blockDim.x + threadIdx.x;
    long stride = (long)gridDim.x * blockDim.x;
    uint4 z; z.x = z.y = z.z = z.w = 0u;
    for (; i < n; i += stride) p[i] = z;
}

// src (N x K) f32 -> dst (Np x Kp) bf16, zero padded
__global__ void k_convert_pad(const float* __restrict__ src, unsigned short* __restrict__ dst,
                              int N, int K, int Np, int Kp) {
    long total = (long)Np * Kp;
    long i = (long)blockIdx.x * blockDim.x + threadIdx.x;
    long stride = (long)gridDim.x * blockDim.x;
    for (; i < total; i += stride) {
        int n = (int)(i / Kp), k = (int)(i % Kp);
        dst[i] = (n < N && k < K) ? f2bf(src[(size_t)n * K + k]) : (unsigned short)0;
    }
}

__global__ void k_gbias(const float* b0, const float* c0, const float* b1, const float* c1,
                        const float* b2, const float* c2, float* gb) {
    int i = blockIdx.x * 256 + threadIdx.x;
    if (i >= 4800) return;
    int l = i / 1600, g = i % 1600;
    const float* x = (l == 0) ? b0 : ((l == 1) ? b1 : b2);
    const float* y = (l == 0) ? c0 : ((l == 1) ? c1 : c2);
    gb[i] = x[g] + y[g];
}

// strokes (B,800,3) -> xin time-major cols 0..2 (bf16)
__global__ void k_strokes_xin(const float* __restrict__ strokes, unsigned short* __restrict__ xin) {
    int i = blockIdx.x * 256 + threadIdx.x;
    if (i >= 64 * 800 * 3) return;
    int t = i / 192, rem = i % 192, b = rem / 3, c = rem % 3;
    xin[((size_t)t * 64 + b) * 544 + c] = f2bf(strokes[((size_t)b * 800 + t) * 3 + c]);
}

// xg0[t*64+b][g] = gb0[g] + sum_i strokes[b,t,i]*Wih0[g,i]
__global__ void k_xg0(const float* __restrict__ strokes, const float* __restrict__ Wih0,
                      const float* __restrict__ gb, float* __restrict__ xg) {
    long i = (long)blockIdx.x * 256 + threadIdx.x;
    if (i >= 51200L * 1600) return;
    int g = (int)(i % 1600);
    long row = i / 1600;
    int t = (int)(row >> 6), b = (int)(row & 63);
    const float* sp = strokes + ((size_t)b * 800 + t) * 3;
    xg[i] = gb[g] + sp[0] * Wih0[g * 3] + sp[1] * Wih0[g * 3 + 1] + sp[2] * Wih0[g * 3 + 2];
}

// hbuf[0][m][col] = h_0[layer][col] (broadcast over batch)
__global__ void k_fill(const float* __restrict__ h0row, unsigned short* __restrict__ hbuf) {
    int i = blockIdx.x * 256 + threadIdx.x;
    if (i >= 64 * 400) return;
    int m = i / 400, col = i % 400;
    hbuf[m * 416 + col] = f2bf(h0row[col]);
}

// ---------------------------------------------------------------- generic WMMA GEMM
// C(M x Nstore) = A(M x Kp)bf16 @ Bt(Np x Kp)bf16^T + bias ; 128x64 workgroup tile.
__global__ void __launch_bounds__(256) k_gemm(const unsigned short* __restrict__ A,
                                              const unsigned short* __restrict__ Bt,
                                              const float* __restrict__ bias,
                                              float* __restrict__ C,
                                              int M, int Kp, int Np, int Nbias,
                                              int Nstore, int ldc, int mode) {
    __shared__ unsigned short As[128 * 40];
    __shared__ unsigned short Bs[64 * 40];
    int tid = threadIdx.x, lane = tid & 31, wv = tid >> 5;
    int wm = wv >> 1, wn = wv & 1;
    int rowBase = blockIdx.y * 128, n0 = blockIdx.x * 64;
    int ln = lane & 15, kh = lane >> 4;
    int arow = tid >> 1, ahalf = tid & 1;
    v8f acc[2][2] = {};

    for (int k0 = 0; k0 < Kp; k0 += 32) {
        __syncthreads();
        {
            const unsigned short* src = A + (size_t)(rowBase + arow) * Kp + k0 + ahalf * 16;
            const uint4* s = (const uint4*)src;
            uint4* d = (uint4*)(As + arow * 40 + ahalf * 16);
            d[0] = s[0]; d[1] = s[1];
            if (k0 + 32 < Kp) __builtin_prefetch(src + 32, 0, 1);
        }
        if (tid < 128) {
            int brow = tid >> 1, bhalf = tid & 1;
            uint4* d = (uint4*)(Bs + brow * 40 + bhalf * 16);
            if (n0 + brow < Np) {
                const uint4* s = (const uint4*)(Bt + (size_t)(n0 + brow) * Kp + k0 + bhalf * 16);
                d[0] = s[0]; d[1] = s[1];
            } else {
                uint4 z; z.x = z.y = z.z = z.w = 0u;
                d[0] = z; d[1] = z;
            }
        }
        __syncthreads();
        FragBF a[2], b[2];
#pragma unroll
        for (int f = 0; f < 2; ++f) {
            const unsigned short* pa = As + (wm * 32 + f * 16 + ln) * 40 + kh * 8;
            a[f].q[0] = *(const uint4*)pa; a[f].q[1] = *(const uint4*)(pa + 16);
            const unsigned short* pb = Bs + (wn * 32 + f * 16 + ln) * 40 + kh * 8;
            b[f].q[0] = *(const uint4*)pb; b[f].q[1] = *(const uint4*)(pb + 16);
        }
#pragma unroll
        for (int fm = 0; fm < 2; ++fm)
#pragma unroll
            for (int fn = 0; fn < 2; ++fn)
                acc[fm][fn] = bfwmma(a[fm], b[fn], acc[fm][fn]);
    }

#pragma unroll
    for (int fm = 0; fm < 2; ++fm)
#pragma unroll
        for (int fn = 0; fn < 2; ++fn) {
            int n = n0 + wn * 32 + fn * 16 + ln;
            if (n >= Nstore) continue;
            float bv = (n < Nbias) ? bias[n] : 0.f;
#pragma unroll
            for (int r = 0; r < 8; ++r) {
                int m = rowBase + wm * 32 + fm * 16 + r + 8 * kh;
                if (m >= M) continue;
                float val = acc[fm][fn][r] + bv;
                if (mode == 0) C[(size_t)m * ldc + n] = val;
                else { int t = m >> 6, bb = m & 63; C[((size_t)bb * 800 + t) * 121 + n] = val; }
            }
        }
}

// ---------------------------------------------------------------- persistent LSTM
__device__ inline void grid_sync(unsigned* cnt, unsigned target) {
    __builtin_amdgcn_fence(__ATOMIC_RELEASE, "agent");
    __syncthreads();
    if (threadIdx.x == 0) {
        __hip_atomic_fetch_add(cnt, 1u, __ATOMIC_RELAXED, __HIP_MEMORY_SCOPE_AGENT);
        while (__hip_atomic_load(cnt, __ATOMIC_RELAXED, __HIP_MEMORY_SCOPE_AGENT) < target)
            __builtin_amdgcn_s_sleep(2);
    }
    __syncthreads();
    __builtin_amdgcn_fence(__ATOMIC_ACQUIRE, "agent");
}

// 25 workgroups; wg owns h-cols [wg*16, wg*16+16), gate cols q*400+wg*16 for q=0..3.
__global__ void __launch_bounds__(256) k_lstm(const float* __restrict__ xg,
                                              const unsigned short* __restrict__ Whhb,  // 1600 x 416
                                              const float* __restrict__ h0row,          // 800 floats
                                              unsigned short* __restrict__ hbuf,        // 2 x 64 x 416
                                              unsigned short* __restrict__ hcat, int hcat_off,
                                              unsigned short* __restrict__ hs0p,
                                              unsigned short* __restrict__ xin,
                                              unsigned* cnt) {
    __shared__ unsigned short Bs[64 * 416];
    __shared__ float Gs[4 * 64 * 16];
    int tid = threadIdx.x, lane = tid & 31, wv = tid >> 5, wg = blockIdx.x;

    // ---- stage Whh slice into LDS once.
    // Rows n = q*16+jj map to gate col q*400 + wg*16 + jj; per gate block the 16
    // rows (16*416 bf16 = 13312B) are contiguous in Whhb, so the whole 53KB slice
    // is a 2D tile: tile_dim0 = 6656 elems, tile_dim1 = 4, stride = 400*416 elems.
#if USE_TDM
    if (tid < 32) {
        unsigned lds = (unsigned)(size_t)(void*)Bs;
        unsigned long long ga = (unsigned long long)(size_t)(Whhb + (size_t)wg * 16 * 416);
        v4u g0;
        g0[0] = 1u;                                               // count=1, user descriptor
        g0[1] = lds;                                              // lds_addr
        g0[2] = (unsigned)ga;                                     // global_addr[31:0]
        g0[3] = (unsigned)((ga >> 32) & 0x01FFFFFFu) | 0x80000000u; // addr[56:32] | type=2
        v8i g1;
        g1[0] = 0x00010000;                                       // data_size=1 (2B)
        g1[1] = (int)((6656u & 0xFFFFu) << 16);                   // tensor_dim0[15:0]
        g1[2] = (int)((6656u >> 16) | (4u << 16));                // td0[31:16] | tensor_dim1[15:0]
        g1[3] = (int)(6656u << 16);                               // td1[31:16]=0 | tile_dim0
        g1[4] = 4;                                                // tile_dim1=4, tile_dim2=0
        g1[5] = 166400;                                           // tensor_dim0_stride[31:0]
        g1[6] = 0;                                                // stride0 hi | stride1 lo
        g1[7] = 0;
        v4i z4 = {0, 0, 0, 0};
#if __clang_major__ >= 23
        v8i z8 = {0, 0, 0, 0, 0, 0, 0, 0};
        __builtin_amdgcn_tensor_load_to_lds(g0, g1, z4, z4, z8, 0);
#else
        __builtin_amdgcn_tensor_load_to_lds(g0, g1, z4, z4, 0);
#endif
    }
    __builtin_amdgcn_s_wait_tensorcnt((short)0);
#else
    for (int idx = tid; idx < 64 * 52; idx += 256) {
        int rrow = idx / 52, chunk = idx % 52;
        int ng = (rrow >> 4) * 400 + wg * 16 + (rrow & 15);
        ((uint4*)(Bs + rrow * 416))[chunk] = ((const uint4*)(Whhb + (size_t)ng * 416))[chunk];
    }
#endif
    float cst[4];
#pragma unroll
    for (int q = 0; q < 4; ++q) {
        int cell = tid * 4 + q;
        cst[q] = h0row[400 + wg * 16 + (cell & 15)];
    }
    __syncthreads();

    int mt = wv >> 1, gt0 = (wv & 1) * 2;
    int ln = lane & 15, kh = lane >> 4;

    for (int t = 0; t < 800; ++t) {
        size_t rp = (size_t)(t & 1) * 64 * 416;
        size_t wp = (size_t)((t + 1) & 1) * 64 * 416;
        v8f acc0, acc1;
        int gc0 = gt0 * 400 + wg * 16 + ln;
#pragma unroll
        for (int r = 0; r < 8; ++r) {
            int mr = mt * 16 + r + 8 * kh;
            const float* row = xg + ((size_t)t * 64 + mr) * 1600;
            acc0[r] = row[gc0];
            acc1[r] = row[gc0 + 400];
        }
#pragma unroll
        for (int kc2 = 0; kc2 < 13; ++kc2) {
            int kc = kc2 * 32;
            FragBF a, b0, b1;
            const unsigned short* ha = hbuf + rp + (mt * 16 + ln) * 416 + kc + kh * 8;
            a.q[0] = *(const uint4*)ha; a.q[1] = *(const uint4*)(ha + 16);
            const unsigned short* p0 = Bs + (gt0 * 16 + ln) * 416 + kc + kh * 8;
            b0.q[0] = *(const uint4*)p0; b0.q[1] = *(const uint4*)(p0 + 16);
            const unsigned short* p1 = Bs + ((gt0 + 1) * 16 + ln) * 416 + kc + kh * 8;
            b1.q[0] = *(const uint4*)p1; b1.q[1] = *(const uint4*)(p1 + 16);
            acc0 = bfwmma(a, b0, acc0);
            acc1 = bfwmma(a, b1, acc1);
        }
#pragma unroll
        for (int r = 0; r < 8; ++r) {
            int mr = mt * 16 + r + 8 * kh;
            Gs[gt0 * 1024 + mr * 16 + ln] = acc0[r];
            Gs[(gt0 + 1) * 1024 + mr * 16 + ln] = acc1[r];
        }
        __syncthreads();
#pragma unroll
        for (int q = 0; q < 4; ++q) {
            int cell = tid * 4 + q;
            int m = cell >> 4, j = cell & 15;
            float ig = Gs[m * 16 + j];
            float fg = Gs[1024 + m * 16 + j];
            float gg = Gs[2048 + m * 16 + j];
            float og = Gs[3072 + m * 16 + j];
            float cc = sigmoidf(fg) * cst[q] + sigmoidf(ig) * tanhf(gg);
            cst[q] = cc;
            float h = sigmoidf(og) * tanhf(cc);
            int hc = wg * 16 + j;
            unsigned short hb = f2bf(h);
            size_t rowi = (size_t)t * 64 + m;
            hbuf[wp + m * 416 + hc] = hb;
            hcat[rowi * 1216 + hcat_off + hc] = hb;
            if (hs0p) hs0p[rowi * 416 + hc] = hb;
            if (xin) xin[rowi * 544 + 3 + hc] = hb;
        }
        grid_sync(cnt, 25u * (unsigned)(t + 1));
    }
}

// ---------------------------------------------------------------- attention
// kappa cumsum: one thread per (b, ma)
__global__ void k_kappa(const float* __restrict__ pbuf, float* __restrict__ kcum) {
    int i = blockIdx.x * 256 + threadIdx.x;
    if (i >= 640) return;
    int b = i / 10, ma = i % 10;
    float k = 0.f;
    for (int t = 0; t < 800; ++t) {
        size_t row = (size_t)t * 64 + b;
        k += 0.2f * __expf(pbuf[row * 32 + 20 + ma]);
        kcum[row * 10 + ma] = k;
    }
}

// per-batch window compute: phi then w -> xin cols 403..530 (bf16)
__global__ void __launch_bounds__(256) k_wctx(const float* __restrict__ context,
                                              const float* __restrict__ cmask,
                                              const float* __restrict__ pbuf,
                                              const float* __restrict__ kcum,
                                              unsigned short* __restrict__ xin) {
    __shared__ float ctx[80 * 128];
    __shared__ float mk[80];
    __shared__ float phi[80];
    __shared__ float am[10], bm[10], km[10];
    int b = blockIdx.x, tid = threadIdx.x;
    for (int i = tid; i < 80 * 128; i += 256) ctx[i] = context[(size_t)b * 10240 + i];
    if (tid < 80) mk[tid] = cmask[b * 80 + tid];
    __syncthreads();
    for (int t = 0; t < 800; ++t) {
        size_t row = (size_t)t * 64 + b;
        if (tid < 10) {
            am[tid] = __expf(pbuf[row * 32 + tid]);
            bm[tid] = __expf(pbuf[row * 32 + 10 + tid]);
            km[tid] = kcum[row * 10 + tid];
        }
        __syncthreads();
        if (tid < 80) {
            float u = (float)tid, s = 0.f;
#pragma unroll
            for (int ma = 0; ma < 10; ++ma) {
                float d = km[ma] - u;
                s += am[ma] * __expf(-bm[ma] * d * d);
            }
            phi[tid] = s * mk[tid];
        }
        __syncthreads();
        if (tid < 128) {
            float s = 0.f;
            for (int u = 0; u < 80; ++u) s += phi[u] * ctx[u * 128 + tid];
            xin[row * 544 + 403 + tid] = f2bf(s);
        }
        __syncthreads();
    }
}

// ---------------------------------------------------------------- host
extern "C" void kernel_launch(void* const* d_in, const int* in_sizes, int n_in,
                              void* d_out, int out_size, void* d_ws, size_t ws_size,
                              hipStream_t stream) {
    (void)in_sizes; (void)n_in; (void)out_size; (void)ws_size;
    const float* strokes = (const float*)d_in[0];
    const float* context = (const float*)d_in[1];
    const float* cmask   = (const float*)d_in[2];
    const float* h_0     = (const float*)d_in[3];
    const float* Wih0    = (const float*)d_in[4];
    const float* Whh0    = (const float*)d_in[5];
    const float* bih0    = (const float*)d_in[6];
    const float* bhh0    = (const float*)d_in[7];
    const float* Wih1    = (const float*)d_in[8];
    const float* Whh1    = (const float*)d_in[9];
    const float* bih1    = (const float*)d_in[10];
    const float* bhh1    = (const float*)d_in[11];
    const float* Wih2    = (const float*)d_in[12];
    const float* Whh2    = (const float*)d_in[13];
    const float* bih2    = (const float*)d_in[14];
    const float* bhh2    = (const float*)d_in[15];
    const float* attn_W  = (const float*)d_in[16];
    const float* attn_b  = (const float*)d_in[17];
    const float* out_W   = (const float*)d_in[18];
    const float* out_b   = (const float*)d_in[19];

    char* w = (char*)d_ws;
    auto alloc = [&](size_t bytes) { char* p = w; w += (bytes + 255) & ~(size_t)255; return p; };

    // zero-initialized region (must be first / contiguous)
    unsigned short* XIN  = (unsigned short*)alloc((size_t)51200 * 544 * 2);   // [strokes|h|w] bf16
    unsigned short* HS0P = (unsigned short*)alloc((size_t)51200 * 416 * 2);   // h0s bf16, K-padded
    unsigned short* HCAT = (unsigned short*)alloc((size_t)51200 * 1216 * 2);  // [h0|h1|h2] bf16
    unsigned short* HBUF = (unsigned short*)alloc((size_t)2 * 64 * 416 * 2);  // ping-pong h
    unsigned* CNT        = (unsigned*)alloc(256);
    size_t zero_bytes = (size_t)(w - (char*)d_ws);

    float* XG            = (float*)alloc((size_t)51200 * 1600 * 4);
    float* PBUF          = (float*)alloc((size_t)51200 * 32 * 4);
    float* KCUM          = (float*)alloc((size_t)51200 * 10 * 4);
    unsigned short* WHHB0 = (unsigned short*)alloc((size_t)1600 * 416 * 2);
    unsigned short* WHHB1 = (unsigned short*)alloc((size_t)1600 * 416 * 2);
    unsigned short* WHHB2 = (unsigned short*)alloc((size_t)1600 * 416 * 2);
    unsigned short* W1B   = (unsigned short*)alloc((size_t)1600 * 544 * 2);
    unsigned short* W2B   = (unsigned short*)alloc((size_t)1600 * 544 * 2);
    unsigned short* ATTNB = (unsigned short*)alloc((size_t)32 * 416 * 2);
    unsigned short* OUTWB = (unsigned short*)alloc((size_t)128 * 1216 * 2);
    float* GB             = (float*)alloc((size_t)3 * 1600 * 4);

    // init + weight prep
    k_zero<<<4096, 256, 0, stream>>>((uint4*)d_ws, (long)(zero_bytes / 16));
    k_convert_pad<<<2048, 256, 0, stream>>>(Whh0, WHHB0, 1600, 400, 1600, 416);
    k_convert_pad<<<2048, 256, 0, stream>>>(Whh1, WHHB1, 1600, 400, 1600, 416);
    k_convert_pad<<<2048, 256, 0, stream>>>(Whh2, WHHB2, 1600, 400, 1600, 416);
    k_convert_pad<<<2048, 256, 0, stream>>>(Wih1, W1B, 1600, 531, 1600, 544);
    k_convert_pad<<<2048, 256, 0, stream>>>(Wih2, W2B, 1600, 531, 1600, 544);
    k_convert_pad<<<64, 256, 0, stream>>>(attn_W, ATTNB, 30, 400, 32, 416);
    k_convert_pad<<<640, 256, 0, stream>>>(out_W, OUTWB, 121, 1200, 128, 1216);
    k_gbias<<<19, 256, 0, stream>>>(bih0, bhh0, bih1, bhh1, bih2, bhh2, GB);
    k_strokes_xin<<<600, 256, 0, stream>>>(strokes, XIN);
    k_xg0<<<320000, 256, 0, stream>>>(strokes, Wih0, GB, XG);

    // layer 0
    k_fill<<<100, 256, 0, stream>>>(h_0 + 0, HBUF);
    k_lstm<<<25, 256, 0, stream>>>(XG, WHHB0, h_0 + 0, HBUF, HCAT, 0, HS0P, XIN, CNT + 0);

    // attention: p = h0s @ attn_W^T, kappa cumsum, window -> xin
    k_gemm<<<dim3(1, 400), 256, 0, stream>>>(HS0P, ATTNB, attn_b, PBUF,
                                             51200, 416, 32, 30, 32, 32, 0);
    k_kappa<<<3, 256, 0, stream>>>(PBUF, KCUM);
    k_wctx<<<64, 256, 0, stream>>>(context, cmask, PBUF, KCUM, XIN);

    // layer 1
    k_gemm<<<dim3(25, 400), 256, 0, stream>>>(XIN, W1B, GB + 1600, XG,
                                              51200, 544, 1600, 1600, 1600, 1600, 0);
    k_fill<<<100, 256, 0, stream>>>(h_0 + 800, HBUF);
    k_lstm<<<25, 256, 0, stream>>>(XG, WHHB1, h_0 + 800, HBUF, HCAT, 400, nullptr, XIN, CNT + 1);

    // layer 2
    k_gemm<<<dim3(25, 400), 256, 0, stream>>>(XIN, W2B, GB + 3200, XG,
                                              51200, 544, 1600, 1600, 1600, 1600, 0);
    k_fill<<<100, 256, 0, stream>>>(h_0 + 1600, HBUF);
    k_lstm<<<25, 256, 0, stream>>>(XG, WHHB2, h_0 + 1600, HBUF, HCAT, 800, nullptr, nullptr, CNT + 2);

    // output projection -> d_out (B,TD,121)
    k_gemm<<<dim3(2, 400), 256, 0, stream>>>(HCAT, OUTWB, out_b, (float*)d_out,
                                             51200, 1216, 128, 121, 121, 121, 1);
}
